// MultiHeadAttention_38740605010258
// MI455X (gfx1250) — compile-verified, matrix operand order checked by reading, not run on hardware
//
#include <hip/hip_runtime.h>
#include <stdint.h>

#define D_MODEL   1024
#define NUM_HEADS 16
#define DEPTH     64
#define SEQ       4096
#define BATCH     2
#define MTOT      (BATCH * SEQ)   // 8192 rows for all GEMMs

typedef __attribute__((ext_vector_type(16))) __bf16 v16bf;
typedef __attribute__((ext_vector_type(8)))  float  v8f;

union FragU { v16bf v; uint4 q[2]; };
union U4U   { uint4 q; unsigned short us[8]; };

// ---- feature detection ------------------------------------------------------
#if defined(__has_builtin)
#if __has_builtin(__builtin_amdgcn_tensor_load_to_lds) && __has_builtin(__builtin_amdgcn_s_wait_tensorcnt)
#define CDNA5_HAS_TDM 1
#endif
#endif
#ifndef CDNA5_HAS_TDM
#define CDNA5_HAS_TDM 0
#endif

__device__ __forceinline__ unsigned short f2bf(float f) {
  unsigned int u = __float_as_uint(f);
  u += 0x7FFFu + ((u >> 16) & 1u);   // round-to-nearest-even
  return (unsigned short)(u >> 16);
}

__device__ __forceinline__ uint2 pack4(float4 f) {
  uint2 pk;
  pk.x = (unsigned)f2bf(f.x) | ((unsigned)f2bf(f.y) << 16);
  pk.y = (unsigned)f2bf(f.z) | ((unsigned)f2bf(f.w) << 16);
  return pk;
}

// Per-lane async global->LDS copy of 16 bytes (ASYNCcnt tracked, no VGPR trip).
__device__ __forceinline__ void async_copy_b128(void* ldsDst, const void* gSrc) {
  const unsigned lds = (unsigned)(size_t)ldsDst;   // low 32 bits = LDS offset
  asm volatile("global_load_async_to_lds_b128 %0, %1, off"
               :: "v"(lds), "v"(gSrc)
               : "memory");
}

__device__ __forceinline__ void wait_async0() {
#if defined(__has_builtin) && __has_builtin(__builtin_amdgcn_s_wait_asynccnt)
  __builtin_amdgcn_s_wait_asynccnt(0);
#else
  asm volatile("s_wait_asynccnt 0x0" ::: "memory");
#endif
}

#if CDNA5_HAS_TDM
// One-shot TDM copy of a contiguous run of `nElems` bf16 elements into LDS.
// Builds a 1-D tensor descriptor (D#): count=1, type=2, data_size=2B,
// tensor_dim0 = tile_dim0 = nElems, tensor_dim1 = tile_dim1 = 1.
__device__ __forceinline__ void tdm_load_1d(void* ldsDst, const void* gSrc,
                                            unsigned nElems) {
  typedef unsigned int u32x4 __attribute__((ext_vector_type(4)));
  typedef int          i32x4 __attribute__((ext_vector_type(4)));
  typedef int          i32x8 __attribute__((ext_vector_type(8)));
  const unsigned           lds = (unsigned)(size_t)ldsDst;
  const unsigned long long ga  = (unsigned long long)gSrc;

  u32x4 g0;
  g0[0] = 1u;                                   // count=1 (valid user D#)
  g0[1] = lds;                                  // lds_addr
  g0[2] = (unsigned)(ga & 0xFFFFFFFFu);         // global_addr[31:0]
  g0[3] = (unsigned)((ga >> 32) & 0x01FFFFFFu)  // global_addr[56:32]
        | (2u << 30);                           // type = 2 ("image")
  i32x8 g1;
  g1[0] = 0x00010000;                           // data_size=1 -> 2 bytes
  g1[1] = (int)((nElems & 0xFFFFu) << 16);      // tensor_dim0[15:0]
  g1[2] = (int)((nElems >> 16) & 0xFFFFu) | (1 << 16);  // dim0[31:16], dim1=1
  g1[3] = (int)((nElems & 0xFFFFu) << 16);      // tile_dim0
  g1[4] = 1;                                    // tile_dim1 = 1
  g1[5] = (int)nElems;                          // tensor_dim0_stride[31:0]
  g1[6] = 0;
  g1[7] = 0;
  i32x4 z4 = {0, 0, 0, 0};
#if __clang_major__ >= 23
  i32x8 z8 = {0, 0, 0, 0, 0, 0, 0, 0};
  __builtin_amdgcn_tensor_load_to_lds(g0, g1, z4, z4, z8, 0);
#else
  __builtin_amdgcn_tensor_load_to_lds(g0, g1, z4, z4, 0);
#endif
}
#endif

// Load one 16x32 bf16 A/B fragment from an LDS tile (row-major, `stride` elems).
//   lanes 0-15 : row = lane,    K chunks {0..7, 16..23}
//   lanes 16-31: row = lane-16, K chunks {8..15, 24..31}
__device__ __forceinline__ v16bf load_frag(const unsigned short* p0, int stride, int lane) {
  const int r = lane & 15;
  const int h = lane >> 4;
  const unsigned short* p = p0 + r * stride + h * 8;
  FragU f;
  f.q[0] = *(const uint4*)(p);
  f.q[1] = *(const uint4*)(p + 16);
  return f.v;
}

__device__ __forceinline__ float red_max16(float v) {
  v = fmaxf(v, __shfl_xor(v, 1, 32));
  v = fmaxf(v, __shfl_xor(v, 2, 32));
  v = fmaxf(v, __shfl_xor(v, 4, 32));
  v = fmaxf(v, __shfl_xor(v, 8, 32));
  return v;
}
__device__ __forceinline__ float red_sum16(float v) {
  v += __shfl_xor(v, 1, 32);
  v += __shfl_xor(v, 2, 32);
  v += __shfl_xor(v, 4, 32);
  v += __shfl_xor(v, 8, 32);
  return v;
}

// ---------------------------------------------------------------------------
// fp32 -> bf16 conversion (weights), 4 elements / thread
// ---------------------------------------------------------------------------
__global__ __launch_bounds__(256)
void cvt_f32_bf16(const float* __restrict__ src, unsigned short* __restrict__ dst) {
  const int i = (blockIdx.x * 256 + threadIdx.x) * 4;
  const float4 f = *(const float4*)(src + i);
  *(uint2*)(dst + i) = pack4(f);
}

// ---------------------------------------------------------------------------
// GEMM: C[m,n] = sum_k A[m,k] * W[n,k] + bias[n]
//   A: fp32 (ABF16==0) or bf16 (ABF16==1), row-major M x 1024
//   W: bf16, 1024 x 1024 row-major (n-major == B-operand layout)
//   OUTMODE 0: bf16 scattered to (B,H,S,DEPTH);  1: fp32 contiguous M x N
// 128 threads = 4 waves (2x2), block tile 128x128, double-buffered LDS,
// async global->LDS staging overlapped with WMMA compute.
// ---------------------------------------------------------------------------
template <int ABF16, int OUTMODE>
__global__ __launch_bounds__(128)
void gemm_wmma_bias(const void* __restrict__ Ap,
                    const unsigned short* __restrict__ Wb,
                    const float* __restrict__ bias, void* __restrict__ out)
{
  __shared__ unsigned short ldsA[2][128 * 32];
  __shared__ unsigned short ldsB[2][128 * 32];

  const int tid  = threadIdx.x;
  const int lane = tid & 31;
  const int wid  = tid >> 5;
  const int wm   = wid >> 1;
  const int wn   = wid & 1;
  const int rowBase = blockIdx.x * 128;
  const int colBase = blockIdx.y * 128;
  const int K = D_MODEL;

  v8f acc[4][4];
  #pragma unroll
  for (int i = 0; i < 4; ++i)
    #pragma unroll
    for (int j = 0; j < 4; ++j)
      acc[i][j] = v8f{0.f, 0.f, 0.f, 0.f, 0.f, 0.f, 0.f, 0.f};

  // ---- prologue: stage k0 = 0 into buffer 0
  {
    const unsigned short* wsrc = Wb + (size_t)(colBase + tid) * K;
    #pragma unroll
    for (int c = 0; c < 4; ++c)
      async_copy_b128(&ldsB[0][tid * 32 + c * 8], wsrc + c * 8);
    if (ABF16) {
      const unsigned short* asrc = (const unsigned short*)Ap + (size_t)(rowBase + tid) * K;
      #pragma unroll
      for (int c = 0; c < 4; ++c)
        async_copy_b128(&ldsA[0][tid * 32 + c * 8], asrc + c * 8);
    } else {
      const float* ap = (const float*)Ap + (size_t)(rowBase + tid) * K;
      #pragma unroll
      for (int j = 0; j < 8; ++j)
        *(uint2*)&ldsA[0][tid * 32 + j * 4] = pack4(((const float4*)ap)[j]);
    }
    wait_async0();
    __syncthreads();
  }

  for (int k0 = 0; k0 < K; k0 += 32) {
    const int cb = (k0 >> 5) & 1;
    const int nb = cb ^ 1;
    const int kn = k0 + 32;

    // ---- issue staging for next K tile (overlaps with compute below)
    float4 areg[8];
    if (kn < K) {
      const unsigned short* wsrc = Wb + (size_t)(colBase + tid) * K + kn;
      #pragma unroll
      for (int c = 0; c < 4; ++c)
        async_copy_b128(&ldsB[nb][tid * 32 + c * 8], wsrc + c * 8);
      if (ABF16) {
        const unsigned short* asrc =
            (const unsigned short*)Ap + (size_t)(rowBase + tid) * K + kn;
        #pragma unroll
        for (int c = 0; c < 4; ++c)
          async_copy_b128(&ldsA[nb][tid * 32 + c * 8], asrc + c * 8);
      } else {
        const float* ap = (const float*)Ap + (size_t)(rowBase + tid) * K + kn;
        #pragma unroll
        for (int j = 0; j < 8; ++j) areg[j] = ((const float4*)ap)[j];
      }
    }

    // ---- compute on current buffer
    v16bf a[4], b[4];
    #pragma unroll
    for (int i = 0; i < 4; ++i)
      a[i] = load_frag(&ldsA[cb][(wm * 64 + i * 16) * 32], 32, lane);
    #pragma unroll
    for (int j = 0; j < 4; ++j)
      b[j] = load_frag(&ldsB[cb][(wn * 64 + j * 16) * 32], 32, lane);
    #pragma unroll
    for (int i = 0; i < 4; ++i)
      #pragma unroll
      for (int j = 0; j < 4; ++j)
        acc[i][j] = __builtin_amdgcn_wmma_f32_16x16x32_bf16(
            false, a[i], false, b[j], (short)0, acc[i][j], false, false);

    if (!ABF16 && kn < K) {
      #pragma unroll
      for (int j = 0; j < 8; ++j)
        *(uint2*)&ldsA[nb][tid * 32 + j * 4] = pack4(areg[j]);
    }
    wait_async0();
    __syncthreads();
  }

  // ---- epilogue: bias add + store
  #pragma unroll
  for (int i = 0; i < 4; ++i) {
    const int mrow = rowBase + wm * 64 + i * 16 + ((lane >> 4) << 3);
    #pragma unroll
    for (int j = 0; j < 4; ++j) {
      const int ncol = colBase + wn * 64 + j * 16 + (lane & 15);
      const float bv = bias[ncol];
      #pragma unroll
      for (int r = 0; r < 8; ++r) {
        const float val = acc[i][j][r] + bv;
        const int grow = mrow + r;
        if (OUTMODE == 0) {
          const int h = ncol >> 6, d = ncol & 63;
          const int bb = grow >> 12, s = grow & 4095;   // SEQ = 4096
          ((unsigned short*)out)[(((size_t)bb * NUM_HEADS + h) * SEQ + s) * DEPTH + d] =
              f2bf(val);
        } else {
          ((float*)out)[(size_t)grow * D_MODEL + ncol] = val;
        }
      }
    }
  }
}

// ---------------------------------------------------------------------------
// Flash attention: block = 128 threads = 4 waves; each wave owns 32 queries.
// Q tile + K tiles staged with the Tensor Data Mover (fallback: async-to-LDS),
// K/Vt/mask double-buffered; next tile staged while current tile computes.
// q/k/v: bf16 (B,H,S,DEPTH).  ctx out: bf16 (B,S,D_MODEL) merged heads.
// ---------------------------------------------------------------------------
__global__ __launch_bounds__(128)
void flash_attn(const unsigned short* __restrict__ q,
                const unsigned short* __restrict__ k,
                const unsigned short* __restrict__ v,
                const float* __restrict__ mask,
                unsigned short* __restrict__ ctx)
{
  __shared__ unsigned short ldsQ[128 * 64];          // 16 KB
  __shared__ unsigned short ldsK[2][64 * 64];        // 16 KB (row=key, col=d)
  __shared__ unsigned short ldsVt[2][64 * 64];       // 16 KB (row=d, col=key)
  __shared__ unsigned short ldsP[4][32 * 64];        // 16 KB (per-wave P)
  __shared__ float          ldsMask[2][64];

  const int tid  = threadIdx.x;
  const int lane = tid & 31;
  const int wave = tid >> 5;
  const int bh   = blockIdx.y;
  const int b    = bh >> 4;          // / NUM_HEADS
  const int h    = bh & 15;
  const int qBase = blockIdx.x * 128;
  const size_t headOff = (size_t)bh * SEQ * DEPTH;
  const unsigned short* qh = q + headOff;
  const unsigned short* kh = k + headOff;
  const unsigned short* vh = v + headOff;
  const float* mp = mask + (size_t)b * SEQ;

  // K tile = 64 contiguous rows of 64 bf16 => one contiguous 8 KB block.
  auto stageK = [&](int buf, int keyBase) {
#if CDNA5_HAS_TDM
    if (wave == 0)
      tdm_load_1d(&ldsK[buf][0], kh + (size_t)keyBase * DEPTH, 64 * DEPTH);
#else
    const unsigned short* src = kh + (size_t)keyBase * DEPTH;
    #pragma unroll
    for (int c = 0; c < 4; ++c)
      async_copy_b128(&ldsK[buf][(tid * 4 + c) * 8], src + (tid * 4 + c) * 8);
#endif
  };
  auto endStageWait = [&]() {
#if CDNA5_HAS_TDM
    if (wave == 0) __builtin_amdgcn_s_wait_tensorcnt(0);
#else
    wait_async0();
#endif
  };
  auto loadV = [&](U4U* vr, int keyBase) {
    #pragma unroll
    for (int c = 0; c < 4; ++c) {
      const int idx = tid * 4 + c;
      const int key = idx >> 3, d0 = (idx & 7) * 8;
      vr[c].q = *(const uint4*)(vh + (size_t)(keyBase + key) * DEPTH + d0);
    }
  };
  auto scatterV = [&](const U4U* vr, int buf) {
    #pragma unroll
    for (int c = 0; c < 4; ++c) {
      const int idx = tid * 4 + c;
      const int key = idx >> 3, d0 = (idx & 7) * 8;
      #pragma unroll
      for (int e = 0; e < 8; ++e) ldsVt[buf][(d0 + e) * 64 + key] = vr[c].us[e];
    }
  };

  // ---- stage Q tile (contiguous 16 KB) once
#if CDNA5_HAS_TDM
  if (wave == 0) {
    tdm_load_1d(&ldsQ[0], qh + (size_t)qBase * DEPTH, 128 * DEPTH);
    __builtin_amdgcn_s_wait_tensorcnt(0);
  }
#else
  {
    const unsigned short* src = qh + (size_t)qBase * DEPTH;
    #pragma unroll
    for (int c = 0; c < 8; ++c)
      async_copy_b128(&ldsQ[(tid * 8 + c) * 8], src + (tid * 8 + c) * 8);
    wait_async0();
  }
#endif
  __syncthreads();

  // q A-fragments: kept in registers for the whole kernel
  v16bf aq[2][2];
  #pragma unroll
  for (int mi = 0; mi < 2; ++mi)
    #pragma unroll
    for (int ks = 0; ks < 2; ++ks)
      aq[mi][ks] = load_frag(&ldsQ[(wave * 32 + mi * 16) * 64 + ks * 32], 64, lane);

  v8f acc[2][4];
  #pragma unroll
  for (int mi = 0; mi < 2; ++mi)
    #pragma unroll
    for (int nd = 0; nd < 4; ++nd)
      acc[mi][nd] = v8f{0.f, 0.f, 0.f, 0.f, 0.f, 0.f, 0.f, 0.f};

  float mstate[2][8], lstate[2][8];
  #pragma unroll
  for (int mi = 0; mi < 2; ++mi)
    #pragma unroll
    for (int r = 0; r < 8; ++r) { mstate[mi][r] = -3.0e38f; lstate[mi][r] = 0.f; }

  const float scale = 0.125f;   // 1/sqrt(64)
  const int NT = SEQ / 64;

  // ---- prologue: stage key tile 0 into buffer 0
  {
    stageK(0, 0);
    U4U vr[4];
    loadV(vr, 0);
    if (tid < 64) ldsMask[0][tid] = mp[tid];
    scatterV(vr, 0);
    endStageWait();
    __syncthreads();
  }

  for (int kt = 0; kt < NT; ++kt) {
    const int cb = kt & 1;
    const int nb = cb ^ 1;
    const bool more = (kt + 1) < NT;

    // ---- issue staging for next key tile (overlaps with compute)
    U4U vr[4];
    if (more) {
      const int keyNext = (kt + 1) * 64;
      stageK(nb, keyNext);
      loadV(vr, keyNext);
      if (tid < 64) ldsMask[nb][tid] = mp[keyNext + tid];
    }

    // ---- scores: S = q . K^T   (M=32, N=64 keys, K=64 depth)
    v8f sc[2][4];
    #pragma unroll
    for (int mi = 0; mi < 2; ++mi)
      #pragma unroll
      for (int nj = 0; nj < 4; ++nj)
        sc[mi][nj] = v8f{0.f, 0.f, 0.f, 0.f, 0.f, 0.f, 0.f, 0.f};

    #pragma unroll
    for (int ks = 0; ks < 2; ++ks) {
      v16bf bk[4];
      #pragma unroll
      for (int nj = 0; nj < 4; ++nj)
        bk[nj] = load_frag(&ldsK[cb][(nj * 16) * 64 + ks * 32], 64, lane);
      #pragma unroll
      for (int mi = 0; mi < 2; ++mi)
        #pragma unroll
        for (int nj = 0; nj < 4; ++nj)
          sc[mi][nj] = __builtin_amdgcn_wmma_f32_16x16x32_bf16(
              false, aq[mi][ks], false, bk[nj], (short)0, sc[mi][nj], false, false);
    }

    // ---- online softmax (row reductions within 16-lane groups)
    #pragma unroll
    for (int mi = 0; mi < 2; ++mi) {
      #pragma unroll
      for (int r = 0; r < 8; ++r) {
        float xs[4];
        float vmax = -3.0e38f;
        #pragma unroll
        for (int nj = 0; nj < 4; ++nj) {
          float x = sc[mi][nj][r] * scale;
          const float mv = ldsMask[cb][nj * 16 + (lane & 15)];
          x = (mv == 0.f) ? -1.0e9f : x;
          xs[nj] = x;
          vmax = fmaxf(vmax, x);
        }
        vmax = red_max16(vmax);
        const float mOld  = mstate[mi][r];
        const float mNew  = fmaxf(mOld, vmax);
        const float alpha = __expf(mOld - mNew);
        mstate[mi][r] = mNew;

        float rsum = 0.f;
        const int prow = mi * 16 + ((lane >> 4) << 3) + r;
        #pragma unroll
        for (int nj = 0; nj < 4; ++nj) {
          const float p = __expf(xs[nj] - mNew);
          rsum += p;
          ldsP[wave][prow * 64 + nj * 16 + (lane & 15)] = f2bf(p);
        }
        rsum = red_sum16(rsum);
        lstate[mi][r] = lstate[mi][r] * alpha + rsum;
        #pragma unroll
        for (int nd = 0; nd < 4; ++nd) acc[mi][nd][r] *= alpha;
      }
    }

    // ---- context accumulate: acc += P . V  (contraction over 64 keys)
    #pragma unroll
    for (int ks = 0; ks < 2; ++ks) {
      v16bf bvf[4];
      #pragma unroll
      for (int nd = 0; nd < 4; ++nd)
        bvf[nd] = load_frag(&ldsVt[cb][(nd * 16) * 64 + ks * 32], 64, lane);
      #pragma unroll
      for (int mi = 0; mi < 2; ++mi) {
        const v16bf ap = load_frag(&ldsP[wave][(mi * 16) * 64 + ks * 32], 64, lane);
        #pragma unroll
        for (int nd = 0; nd < 4; ++nd)
          acc[mi][nd] = __builtin_amdgcn_wmma_f32_16x16x32_bf16(
              false, ap, false, bvf[nd], (short)0, acc[mi][nd], false, false);
      }
    }

    // ---- finish staging of next tile, then swap buffers
    if (more) scatterV(vr, nb);
    endStageWait();
    __syncthreads();
  }

  // ---- finalize: divide by row sums, write merged-head ctx (B,S,D_MODEL) bf16
  #pragma unroll
  for (int mi = 0; mi < 2; ++mi) {
    #pragma unroll
    for (int r = 0; r < 8; ++r) {
      const float inv = 1.f / lstate[mi][r];
      const int s = qBase + wave * 32 + mi * 16 + ((lane >> 4) << 3) + r;
      #pragma unroll
      for (int nd = 0; nd < 4; ++nd) {
        const int d = nd * 16 + (lane & 15);
        ctx[((size_t)b * SEQ + s) * D_MODEL + h * DEPTH + d] =
            f2bf(acc[mi][nd][r] * inv);
      }
    }
  }
}

// ---------------------------------------------------------------------------
extern "C" void kernel_launch(void* const* d_in, const int* in_sizes, int n_in,
                              void* d_out, int out_size, void* d_ws, size_t ws_size,
                              hipStream_t stream)
{
  (void)in_sizes; (void)n_in; (void)out_size; (void)ws_size;

  const float* Q    = (const float*)d_in[0];
  const float* Kin  = (const float*)d_in[1];
  const float* Vin  = (const float*)d_in[2];
  const float* mask = (const float*)d_in[3];
  const float* Wq   = (const float*)d_in[4];
  const float* bq   = (const float*)d_in[5];
  const float* Wk   = (const float*)d_in[6];
  const float* bk   = (const float*)d_in[7];
  const float* Wv   = (const float*)d_in[8];
  const float* bv   = (const float*)d_in[9];
  const float* Wo   = (const float*)d_in[10];
  const float* bo   = (const float*)d_in[11];

  // workspace: 4 x (8192*1024) + 4 x (1024*1024) bf16 = 72 MB
  unsigned short* qproj = (unsigned short*)d_ws;
  unsigned short* kproj = qproj + (size_t)MTOT * D_MODEL;
  unsigned short* vproj = kproj + (size_t)MTOT * D_MODEL;
  unsigned short* ctx   = vproj + (size_t)MTOT * D_MODEL;
  unsigned short* Wqb   = ctx   + (size_t)MTOT * D_MODEL;
  unsigned short* Wkb   = Wqb   + (size_t)D_MODEL * D_MODEL;
  unsigned short* Wvb   = Wkb   + (size_t)D_MODEL * D_MODEL;
  unsigned short* Wob   = Wvb   + (size_t)D_MODEL * D_MODEL;

  // convert weights fp32 -> bf16
  const int cvtBlocks = (D_MODEL * D_MODEL) / (256 * 4);
  cvt_f32_bf16<<<cvtBlocks, 256, 0, stream>>>(Wq, Wqb);
  cvt_f32_bf16<<<cvtBlocks, 256, 0, stream>>>(Wk, Wkb);
  cvt_f32_bf16<<<cvtBlocks, 256, 0, stream>>>(Wv, Wvb);
  cvt_f32_bf16<<<cvtBlocks, 256, 0, stream>>>(Wo, Wob);

  dim3 gblock(128);
  dim3 ggrid(MTOT / 128, D_MODEL / 128);     // 64 x 8

  gemm_wmma_bias<0, 0><<<ggrid, gblock, 0, stream>>>(Q,   Wqb, bq, qproj);
  gemm_wmma_bias<0, 0><<<ggrid, gblock, 0, stream>>>(Kin, Wkb, bk, kproj);
  gemm_wmma_bias<0, 0><<<ggrid, gblock, 0, stream>>>(Vin, Wvb, bv, vproj);

  dim3 agrid(SEQ / 128, BATCH * NUM_HEADS);  // 32 x 32
  flash_attn<<<agrid, gblock, 0, stream>>>(qproj, kproj, vproj, mask, ctx);

  gemm_wmma_bias<1, 1><<<ggrid, gblock, 0, stream>>>(ctx, Wob, bo, (float*)d_out);
}
